// DRNNModel_51110110822604
// MI455X (gfx1250) — matrix operand order repaired and checked by reading, not compiled, not run on hardware
//
#include <hip/hip_runtime.h>

#define VOCAB 100
#define HID   1024
#define BATCH 64
#define TLEN  256
#define BT    (BATCH * TLEN)
#define NPAD_D 128
#define KSTEP 64
#define GM    128     // gemm workgroup tile M
#define GN    128     // gemm workgroup tile N

typedef __attribute__((ext_vector_type(8)))  __bf16 bf16x8;
typedef __attribute__((ext_vector_type(16))) __bf16 bf16x16;
typedef __attribute__((ext_vector_type(8)))  float  f32x8;
typedef __attribute__((ext_vector_type(4)))  int    v4i;
typedef __attribute__((ext_vector_type(4)))  unsigned int u32x4;

#if defined(__AMDGCN__) && __has_builtin(__builtin_amdgcn_global_load_async_to_lds_b128) && __has_builtin(__builtin_amdgcn_s_wait_asynccnt)
#define HAVE_ASYNC_LDS 1
#else
#define HAVE_ASYNC_LDS 0
#endif

#if HAVE_ASYNC_LDS
typedef v4i __attribute__((address_space(1)))* gv4i_p;   // global (AS1) int4*
typedef v4i __attribute__((address_space(3)))* lv4i_p;   // LDS (AS3) int4*
#endif

__device__ __forceinline__ void cp_lds_b128(void* lds, const void* g) {
#if HAVE_ASYNC_LDS
  __builtin_amdgcn_global_load_async_to_lds_b128((gv4i_p)g, (lv4i_p)lds, 0, 0);
#else
  *(u32x4*)lds = *(const u32x4*)g;
#endif
}

template <int N>
__device__ __forceinline__ void async_wait() {
#if HAVE_ASYNC_LDS
  __builtin_amdgcn_s_wait_asynccnt(N);
#endif
}

__device__ __forceinline__ __bf16 f2bf(float f) {
  union { float f; unsigned u; } v; v.f = f;
  unsigned r = (v.u + 0x7FFFu + ((v.u >> 16) & 1u)) >> 16;
  union { unsigned short s; __bf16 b; } o; o.s = (unsigned short)r;
  return o.b;
}

__device__ __forceinline__ bf16x16 join16(bf16x8 lo, bf16x8 hi) {
  return __builtin_shufflevector(lo, hi, 0,1,2,3,4,5,6,7,8,9,10,11,12,13,14,15);
}

// A fragment 16x32 bf16 from row-major tile (lane l -> row l&15; K chunks per ISA layout)
__device__ __forceinline__ bf16x16 frag_a(const __bf16* base, int lda, int k0, int lane) {
  const __bf16* row = base + (lane & 15) * lda;
  const int kh = (lane >> 4) * 8;
  bf16x8 lo = *(const bf16x8*)(row + k0 + kh);
  bf16x8 hi = *(const bf16x8*)(row + k0 + 16 + kh);
  return join16(lo, hi);
}

// B fragment 32x16 bf16 from n-major (pre-transposed) tile (lane l -> col l&15; K chunks per ISA layout)
__device__ __forceinline__ bf16x16 frag_b(const __bf16* base, int ldb, int k0, int lane) {
  const __bf16* row = base + (lane & 15) * ldb + k0 + (lane >> 4) * 8;
  bf16x8 lo = *(const bf16x8*)(row);
  bf16x8 hi = *(const bf16x8*)(row + 16);
  return join16(lo, hi);
}

__device__ __forceinline__ f32x8 wmma_bf16(bf16x16 a, bf16x16 b, f32x8 c) {
  return __builtin_amdgcn_wmma_f32_16x16x32_bf16(false, a, false, b, (short)0, c, false, false);
}

__device__ __forceinline__ void grid_barrier(unsigned* count, volatile unsigned* gen, unsigned ngroups) {
  __threadfence();
  __syncthreads();
  if (threadIdx.x == 0) {
    unsigned g = *gen;
    if (atomicAdd(count, 1u) == ngroups - 1u) {
      *count = 0u;
      __threadfence();
      *gen = g + 1u;
    } else {
      while (*gen == g) __builtin_amdgcn_s_sleep(2);
    }
  }
  __syncthreads();
}

// ---------------------------------------------------------------------------
// Persistent recurrent kernel: 16 workgroups, each owns 64 columns of H.
// 4 waves (one per SIMD32), each computing a 32x32 tile = 4 independent
// WMMA accumulator chains (hides the bf16 WMMA co-exec hazard slots).
// LDS: U-slice [64][1024] bf16 (128KB, staged once) + h state [64][1024] bf16
// (128KB, async-restaged every step) = 256KB of the 320KB WGP LDS.
// ---------------------------------------------------------------------------
__global__ __launch_bounds__(128) void rnn_scan_kernel(
    const float* __restrict__ xw,   // [B,T,H] precomputed input projection (f32)
    const __bf16* __restrict__ Ut,  // [H][H] n-major (transposed) bf16
    __bf16* __restrict__ hbuf,      // [2][B][H] bf16 double buffer (hbuf[0] = h0)
    __bf16* __restrict__ xout,      // [B,T,H] bf16 layer output
    unsigned* __restrict__ bar, int ngroups)
{
  extern __shared__ char smem[];
  __bf16* Bs = (__bf16*)smem;             // [64][HID] U columns slice
  __bf16* As = (__bf16*)smem + 64 * HID;  // [64][HID] h state
  const int tid  = threadIdx.x;
  const int lane = tid & 31;
  const int wave = tid >> 5;              // 0..3 (wave32)
  const int nblock = blockIdx.x;

  {
    const __bf16* src = Ut + (size_t)nblock * 64 * HID;
    for (int idx = tid; idx < (64 * HID) / 8; idx += 128)
      cp_lds_b128(Bs + idx * 8, src + idx * 8);
    async_wait<0>();
  }
  __syncthreads();

  const int wm = wave & 1;                // m half: rows 32*wm
  const int wn = wave >> 1;               // n half: cols 32*wn
  const int kh = lane >> 4;
  const int ln = lane & 15;

  for (int t = 0; t < TLEN; ++t) {
    const __bf16* hprev = hbuf + (size_t)(t & 1) * BATCH * HID;
    __bf16* hnext = hbuf + (size_t)((t + 1) & 1) * BATCH * HID;

    for (int idx = tid; idx < (BATCH * HID) / 8; idx += 128)
      cp_lds_b128(As + idx * 8, hprev + idx * 8);

    f32x8 acc[2][2];
#pragma unroll
    for (int mt = 0; mt < 2; ++mt)
#pragma unroll
      for (int nt = 0; nt < 2; ++nt) {
        const int col = nblock * 64 + wn * 32 + nt * 16 + ln;
#pragma unroll
        for (int r = 0; r < 8; ++r) {
          const int row = wm * 32 + mt * 16 + 8 * kh + r;
          acc[mt][nt][r] = xw[((size_t)row * TLEN + t) * HID + col];
        }
      }
    if (t + 1 < TLEN)  // warm L2/WGP$ for next step's C-init loads
      __builtin_prefetch(&xw[((size_t)(wm * 32 + 8 * kh) * TLEN + t + 1) * HID + nblock * 64 + wn * 32 + ln], 0, 1);

    async_wait<0>();
    __syncthreads();

#pragma unroll 4
    for (int k0 = 0; k0 < HID; k0 += 32) {
      bf16x16 a0 = frag_a(As + (wm * 32) * HID,      HID, k0, lane);
      bf16x16 a1 = frag_a(As + (wm * 32 + 16) * HID, HID, k0, lane);
      bf16x16 b0 = frag_b(Bs + (wn * 32) * HID,      HID, k0, lane);
      bf16x16 b1 = frag_b(Bs + (wn * 32 + 16) * HID, HID, k0, lane);
      acc[0][0] = wmma_bf16(a0, b0, acc[0][0]);
      acc[0][1] = wmma_bf16(a0, b1, acc[0][1]);
      acc[1][0] = wmma_bf16(a1, b0, acc[1][0]);
      acc[1][1] = wmma_bf16(a1, b1, acc[1][1]);
    }

#pragma unroll
    for (int mt = 0; mt < 2; ++mt)
#pragma unroll
      for (int nt = 0; nt < 2; ++nt) {
        const int col = nblock * 64 + wn * 32 + nt * 16 + ln;
#pragma unroll
        for (int r = 0; r < 8; ++r) {
          const int row = wm * 32 + mt * 16 + 8 * kh + r;
          __bf16 v = f2bf(tanhf(acc[mt][nt][r]));
          hnext[(size_t)row * HID + col] = v;
          xout[((size_t)row * TLEN + t) * HID + col] = v;
        }
      }
    grid_barrier(bar, bar + 1, (unsigned)ngroups);
  }
}

// ---------------------------------------------------------------------------
// Batched GEMM: C[M][ldc] = A[M][K](bf16) * Bt[Npad][K]^T(bf16) + bias, f32 out.
// 128x128 workgroup tile, 8 waves x (2x4 tiles) = 8 accumulators per wave,
// double-buffered K=64 stages via async-to-LDS (s_wait_asynccnt pipelining).
// ---------------------------------------------------------------------------
__global__ __launch_bounds__(256) void gemm_tn_kernel(
    const __bf16* __restrict__ A, const __bf16* __restrict__ Bt,
    const float* __restrict__ bias, float* __restrict__ C,
    int Ksz, int ldc, int nmax)
{
  __shared__ __bf16 Asm[2][GM * KSTEP];
  __shared__ __bf16 Bsm[2][GN * KSTEP];
  const int tid = threadIdx.x, lane = tid & 31, wave = tid >> 5;
  const size_t mbase = (size_t)blockIdx.x * GM;
  const size_t nbase = (size_t)blockIdx.y * GN;

  const int wm = wave & 3;          // rows 32*wm .. +32
  const int wn = wave >> 2;         // cols 64*wn .. +64
  const int kh = lane >> 4;
  const int ln = lane & 15;

  f32x8 acc[2][4];
#pragma unroll
  for (int nt = 0; nt < 4; ++nt) {
    const int gn = (int)nbase + wn * 64 + nt * 16 + ln;
    const float bv = (gn < nmax) ? bias[gn] : 0.0f;
#pragma unroll
    for (int mt = 0; mt < 2; ++mt)
#pragma unroll
      for (int r = 0; r < 8; ++r) acc[mt][nt][r] = bv;
  }

  auto stage = [&](int buf, int k0) {
    for (int idx = tid; idx < (GM * KSTEP) / 8; idx += 256) {
      const int row = idx >> 3, ch = (idx & 7) * 8;
      cp_lds_b128(&Asm[buf][row * KSTEP + ch], A + (mbase + row) * (size_t)Ksz + k0 + ch);
      cp_lds_b128(&Bsm[buf][row * KSTEP + ch], Bt + (nbase + row) * (size_t)Ksz + k0 + ch);
    }
  };

  const int nstages = Ksz / KSTEP;
  stage(0, 0);
  for (int s = 0; s < nstages; ++s) {
    const int cur = s & 1;
    if (s + 1 < nstages) { stage(cur ^ 1, (s + 1) * KSTEP); async_wait<8>(); }
    else { async_wait<0>(); }
    __syncthreads();
    const __bf16* Ac = Asm[cur];
    const __bf16* Bc = Bsm[cur];
#pragma unroll
    for (int k0 = 0; k0 < KSTEP; k0 += 32) {
      bf16x16 a0 = frag_a(Ac + (wm * 32) * KSTEP,      KSTEP, k0, lane);
      bf16x16 a1 = frag_a(Ac + (wm * 32 + 16) * KSTEP, KSTEP, k0, lane);
      bf16x16 b[4];
#pragma unroll
      for (int nt = 0; nt < 4; ++nt)
        b[nt] = frag_b(Bc + (wn * 64 + nt * 16) * KSTEP, KSTEP, k0, lane);
#pragma unroll
      for (int nt = 0; nt < 4; ++nt) {
        acc[0][nt] = wmma_bf16(a0, b[nt], acc[0][nt]);
        acc[1][nt] = wmma_bf16(a1, b[nt], acc[1][nt]);
      }
    }
    __syncthreads();
  }

#pragma unroll
  for (int mt = 0; mt < 2; ++mt)
#pragma unroll
    for (int nt = 0; nt < 4; ++nt) {
      const int gn = (int)nbase + wn * 64 + nt * 16 + ln;
      if (gn < nmax) {
#pragma unroll
        for (int r = 0; r < 8; ++r) {
          const size_t m = mbase + wm * 32 + mt * 16 + 8 * kh + r;
          C[m * ldc + gn] = acc[mt][nt][r];
        }
      }
    }
}

// ---------------------------------------------------------------------------
// Prep kernels
// ---------------------------------------------------------------------------
__global__ __launch_bounds__(256) void embed_kernel(
    const int* __restrict__ ids, const float* __restrict__ W1,
    const float* __restrict__ b1, float* __restrict__ xw)
{
  const int pos = blockIdx.x;            // b*T + t
  const int row = ids[pos];
  const float* src = W1 + (size_t)row * HID;
  float* dst = xw + (size_t)pos * HID;
  for (int i = threadIdx.x; i < HID; i += 256)
    dst[i] = src[i] + b1[i];
}

__global__ __launch_bounds__(256) void transpose_bf16_kernel(
    const float* __restrict__ S, __bf16* __restrict__ D, int Ksz, int N, int Npad)
{
  const size_t i = (size_t)blockIdx.x * 256 + threadIdx.x;
  if (i >= (size_t)Npad * Ksz) return;
  const int n = (int)(i / Ksz), k = (int)(i % Ksz);
  const float v = (n < N) ? S[(size_t)k * N + n] : 0.0f;
  D[i] = f2bf(v);
}

__global__ __launch_bounds__(256) void init_kernel(__bf16* hbuf, unsigned* bar) {
  const size_t i = (size_t)blockIdx.x * 256 + threadIdx.x;
  if (i < (size_t)2 * BATCH * HID) hbuf[i] = f2bf(0.0f);
  if (i < 8) bar[i] = 0u;
}

// ---------------------------------------------------------------------------
extern "C" void kernel_launch(void* const* d_in, const int* in_sizes, int n_in,
                              void* d_out, int out_size, void* d_ws, size_t ws_size,
                              hipStream_t stream) {
  (void)in_sizes; (void)n_in; (void)out_size; (void)ws_size;
  const int*   ids = (const int*)d_in[0];
  const float* W1  = (const float*)d_in[1];
  const float* Uw[5] = {(const float*)d_in[2], (const float*)d_in[5], (const float*)d_in[8],
                        (const float*)d_in[11], (const float*)d_in[14]};
  const float* bw[5] = {(const float*)d_in[3], (const float*)d_in[6], (const float*)d_in[9],
                        (const float*)d_in[12], (const float*)d_in[15]};
  const float* Ww[4] = {(const float*)d_in[4], (const float*)d_in[7],
                        (const float*)d_in[10], (const float*)d_in[13]};
  const float* Wd = (const float*)d_in[16];
  const float* bd = (const float*)d_in[17];

  char* ws = (char*)d_ws;
  size_t off = 0;
  auto alloc = [&](size_t bytes) { void* p = ws + off; off = (off + bytes + 255) & ~(size_t)255; return p; };

  __bf16* Ut[5]; for (int i = 0; i < 5; ++i) Ut[i] = (__bf16*)alloc((size_t)HID * HID * 2);
  __bf16* Wt[4]; for (int i = 0; i < 4; ++i) Wt[i] = (__bf16*)alloc((size_t)HID * HID * 2);
  __bf16* Wdt  = (__bf16*)alloc((size_t)NPAD_D * HID * 2);
  float*  xw   = (float*)alloc((size_t)BT * HID * 4);
  __bf16* xbf  = (__bf16*)alloc((size_t)BT * HID * 2);
  __bf16* hbuf = (__bf16*)alloc((size_t)2 * BATCH * HID * 2);
  unsigned* bar = (unsigned*)alloc(256);

  init_kernel<<<(2 * BATCH * HID + 255) / 256, 256, 0, stream>>>(hbuf, bar);
  const int nelm = HID * HID;
  for (int i = 0; i < 5; ++i)
    transpose_bf16_kernel<<<(nelm + 255) / 256, 256, 0, stream>>>(Uw[i], Ut[i], HID, HID, HID);
  for (int i = 0; i < 4; ++i)
    transpose_bf16_kernel<<<(nelm + 255) / 256, 256, 0, stream>>>(Ww[i], Wt[i], HID, HID, HID);
  transpose_bf16_kernel<<<(NPAD_D * HID + 255) / 256, 256, 0, stream>>>(Wd, Wdt, HID, VOCAB, NPAD_D);

  embed_kernel<<<BT, 256, 0, stream>>>(ids, W1, bw[0], xw);

  const size_t rnn_lds = (size_t)2 * 64 * HID * sizeof(__bf16);  // 256 KB of 320 KB WGP LDS
  for (int l = 0; l < 5; ++l) {
    rnn_scan_kernel<<<HID / 64, 128, rnn_lds, stream>>>(xw, Ut[l], hbuf, xbf, bar, HID / 64);
    if (l < 4)
      gemm_tn_kernel<<<dim3(BT / GM, HID / GN), 256, 0, stream>>>(xbf, Wt[l], bw[l + 1], xw, HID, HID, HID);
  }
  gemm_tn_kernel<<<dim3(BT / GM, NPAD_D / GN), 256, 0, stream>>>(xbf, Wdt, bd, (float*)d_out, HID, VOCAB, VOCAB);
}
